// SimpleAttention_50328426774584
// MI455X (gfx1250) — matrix operand order repaired and checked by reading, not compiled, not run on hardware
//
#include <hip/hip_runtime.h>
#include <hip/hip_bf16.h>

typedef __attribute__((ext_vector_type(16))) _Float16 v16h;
typedef __attribute__((ext_vector_type(8)))  float    v8f;
typedef int attn_v4i __attribute__((vector_size(16)));

constexpr int kB   = 8;
constexpr int kS   = 2048;
constexpr int kD   = 512;   // D_IN == D_ATT
constexpr int kRot = 256;   // D_ROT

constexpr int TM = 64, TN = 64, TK = 32;   // projection tile
constexpr int TM2 = 64, TN2 = 128;         // attention GEMM tile (8 waves x 32x32)
constexpr int PITCH   = 40;        // halves per padded LDS row (80B: 16B-aligned rows)
constexpr int PITCH_U = PITCH / 2; // dwords per padded LDS row

__device__ __forceinline__ v8f wmma_f16(const v16h& a, const v16h& b, const v8f& c) {
  return __builtin_amdgcn_wmma_f32_16x16x32_f16(false, a, false, b, (short)0, c,
                                                false, false);
}

union FragU { v16h v; unsigned u[8]; };

// ---- gfx1250 async global->LDS staging (16B per lane), with portable fallback
#if __has_builtin(__builtin_amdgcn_global_load_async_to_lds_b128)
#define ATTN_HAS_ASYNC_LDS 1
#else
#define ATTN_HAS_ASYNC_LDS 0
#endif

__device__ __forceinline__ void cp_async16(void* lds, const void* gsrc) {
#if ATTN_HAS_ASYNC_LDS
  __builtin_amdgcn_global_load_async_to_lds_b128(
      (__attribute__((address_space(1))) attn_v4i*)gsrc,
      (__attribute__((address_space(3))) attn_v4i*)lds,
      0 /*offset*/, 0 /*cpol*/);
#else
  *(float4*)lds = *(const float4*)gsrc;
#endif
}

__device__ __forceinline__ void cp_async_wait() {
#if ATTN_HAS_ASYNC_LDS
  asm volatile("s_wait_asynccnt 0x0" ::: "memory");
#endif
}

// ---- One K-step, wave owns 32x16 of C (two 16x16 tiles stacked in M)
__device__ __forceinline__ void mma_step(const _Float16* As, const _Float16* Bs,
                                         int wr, int wc, int l16, int hi,
                                         v8f& acc0, v8f& acc1) {
  const unsigned* AsU = (const unsigned*)As;
  const unsigned* BsU = (const unsigned*)Bs;
  FragU bf;
  const int ncol = wc * 16 + l16;
#pragma unroll
  for (int j = 0; j < 8; ++j)            // B: K = hi*16 + 2j
    bf.u[j] = BsU[ncol * PITCH_U + hi * 8 + j];
  FragU a0, a1;
  const int mrow = wr * 32 + l16;
#pragma unroll
  for (int r = 0; r < 8; ++r) {          // A: K = hi*16 + 2r
    const int kp = hi * 8 + r;
    a0.u[r] = AsU[mrow * PITCH_U + kp];
    a1.u[r] = AsU[(mrow + 16) * PITCH_U + kp];
  }
  acc0 = wmma_f16(a0.v, bf.v, acc0);
  acc1 = wmma_f16(a1.v, bf.v, acc1);
}

// ---- One K-step, wave owns 32x32 of C (2x2 tiles, A/B fragment reuse)
__device__ __forceinline__ void mma_step2(const _Float16* As, const _Float16* Bs,
                                          int wr, int wc, int l16, int hi,
                                          v8f acc[2][2]) {
  const unsigned* AsU = (const unsigned*)As;
  const unsigned* BsU = (const unsigned*)Bs;
  FragU a[2], bfr[2];
#pragma unroll
  for (int mt = 0; mt < 2; ++mt) {
    const int mrow = wr * 32 + mt * 16 + l16;
#pragma unroll
    for (int r = 0; r < 8; ++r) {
      const int kp = hi * 8 + r;
      a[mt].u[r] = AsU[mrow * PITCH_U + kp];
    }
  }
#pragma unroll
  for (int nt = 0; nt < 2; ++nt) {
    const int ncol = wc * 32 + nt * 16 + l16;
#pragma unroll
    for (int j = 0; j < 8; ++j)
      bfr[nt].u[j] = BsU[ncol * PITCH_U + hi * 8 + j];
  }
#pragma unroll
  for (int mt = 0; mt < 2; ++mt)
#pragma unroll
    for (int nt = 0; nt < 2; ++nt)
      acc[mt][nt] = wmma_f16(a[mt].v, bfr[nt].v, acc[mt][nt]);
}

// ---------------- Projection: Y(f16)[M,512] = X(f32)[M,512] @ W(f32)[512,512] + b
__global__ void __launch_bounds__(256)
attn_proj_kernel(const float* __restrict__ X, const float* __restrict__ W,
                 const float* __restrict__ bias, _Float16* __restrict__ Y) {
  __shared__ __align__(16) _Float16 As[TM * PITCH];
  __shared__ __align__(16) _Float16 Bs[TN * PITCH];
  const int t = threadIdx.x;
  const int n0 = blockIdx.x * TN;
  const int m0 = blockIdx.y * TM;
  const int srow = t >> 2, sk0 = (t & 3) * 8;
  const int lane = t & 31, w = t >> 5;
  const int l16 = lane & 15, hi = lane >> 4;
  const int wr = w >> 2, wc = w & 3;
  v8f acc0 = {}; v8f acc1 = {};
  for (int k0 = 0; k0 < kD; k0 += TK) {
    const float* xs = X + (size_t)(m0 + srow) * kD + k0 + sk0;
    const float* ws = W + (size_t)(k0 + sk0) * kD + n0 + srow;
#pragma unroll
    for (int i = 0; i < 8; ++i) {
      As[srow * PITCH + sk0 + i] = (_Float16)xs[i];
      Bs[srow * PITCH + sk0 + i] = (_Float16)ws[(size_t)i * kD]; // W^T staged
    }
    __syncthreads();
    mma_step(As, Bs, wr, wc, l16, hi, acc0, acc1);
    __syncthreads();
  }
  const int ncol = n0 + wc * 16 + l16;
  const float bv = bias[ncol];
#pragma unroll
  for (int r = 0; r < 8; ++r) {
    const int m = m0 + wr * 32 + hi * 8 + r;
    Y[(size_t)m * kD + ncol]        = (_Float16)(acc0[r] + bv);
    Y[(size_t)(m + 16) * kD + ncol] = (_Float16)(acc1[r] + bv);
  }
}

// ---------------- Rotary (in place on f16 q or k), first kRot dims
__global__ void attn_rotary_kernel(_Float16* __restrict__ x) {
  const int idx = blockIdx.x * blockDim.x + threadIdx.x;
  const int npairs = kB * kS * (kRot / 2);
  if (idx >= npairs) return;
  const int i  = idx % (kRot / 2);
  const int bs = idx / (kRot / 2);
  const int pos = bs % kS;
  const float inv = __powf(10000.f, -(float)(2 * i) / (float)kRot);
  const float ang = (float)pos * inv;
  float c, s;
  __sincosf(ang, &s, &c);   // (sin, cos)
  _Float16* p = x + (size_t)bs * kD + 2 * i;
  const float x1 = (float)p[0], x2 = (float)p[1];
  p[0] = (_Float16)(x1 * c - x2 * s);
  p[1] = (_Float16)(x2 * c + x1 * s);
}

// ---------------- Scores: out(f32)[b,q,k] = (q . k)/sqrt(D), masked
__global__ void __launch_bounds__(256)
attn_scores_kernel(const _Float16* __restrict__ qh, const _Float16* __restrict__ kh,
                   const int* __restrict__ mask, float* __restrict__ out) {
  __shared__ __align__(16) _Float16 As[TM2 * PITCH];
  __shared__ __align__(16) _Float16 Bs[TN2 * PITCH];
  const int t = threadIdx.x;
  const int b  = blockIdx.z;
  const int n0 = blockIdx.x * TN2;  // key position tile
  const int m0 = blockIdx.y * TM2;  // query position tile
  const int lane = t & 31, w = t >> 5;
  const int l16 = lane & 15, hi = lane >> 4;
  const int wr = w >> 2, wc = w & 3;
  const int arow = t >> 2, ak0 = (t & 3) * 8;       // A: 256 chunks of 16B
  const int brow0 = t >> 1, bk0 = (t & 1) * 16;     // B: 2 chunks/thread via rows
  v8f acc[2][2] = {};
  for (int k0 = 0; k0 < kD; k0 += TK) {
    // Async stage Q tile rows (64 x 32 halves)
    cp_async16(&As[arow * PITCH + ak0],
               qh + ((size_t)b * kS + m0 + arow) * kD + k0 + ak0);
    // Async stage K tile rows (128 x 32 halves): 2 x 16B per thread
    cp_async16(&Bs[brow0 * PITCH + bk0],
               kh + ((size_t)b * kS + n0 + brow0) * kD + k0 + bk0);
    cp_async16(&Bs[brow0 * PITCH + bk0 + 8],
               kh + ((size_t)b * kS + n0 + brow0) * kD + k0 + bk0 + 8);
    if (k0 + TK < kD)
      __builtin_prefetch(qh + ((size_t)b * kS + m0 + arow) * kD + k0 + TK + ak0, 0, 0);
    cp_async_wait();
    __syncthreads();
    mma_step2(As, Bs, wr, wc, l16, hi, acc);
    __syncthreads();
  }
  const float scale = 0.044194173824159216f; // 1/sqrt(512)
#pragma unroll
  for (int nt = 0; nt < 2; ++nt) {
    const int kcol = n0 + wc * 32 + nt * 16 + l16;
#pragma unroll
    for (int mt = 0; mt < 2; ++mt) {
#pragma unroll
      for (int r = 0; r < 8; ++r) {
        const int q = m0 + wr * 32 + mt * 16 + hi * 8 + r;
        const size_t o = ((size_t)b * kS + q) * kS + kcol;
        const float s = acc[mt][nt][r] * scale;
        out[o] = (mask[o] == 0) ? -1e-9f : s;
      }
    }
  }
}

// ---------------- Softmax over rows of 2048; f32 in/out + f16 copy for GEMM
__global__ void __launch_bounds__(256)
attn_softmax_kernel(float* __restrict__ wgt, _Float16* __restrict__ ph) {
  __shared__ float red[8];
  const size_t row = blockIdx.x;   // b*S + q
  float* p = wgt + row * (size_t)kS;
  const int t = threadIdx.x, lane = t & 31, wid = t >> 5;
  float vals[8];
  float mx = -3.402823466e38f;
#pragma unroll
  for (int i = 0; i < 8; ++i) { vals[i] = p[t + i * 256]; mx = fmaxf(mx, vals[i]); }
#pragma unroll
  for (int o = 16; o > 0; o >>= 1) mx = fmaxf(mx, __shfl_xor(mx, o, 32));
  if (lane == 0) red[wid] = mx;
  __syncthreads();
  float bm = red[0];
#pragma unroll
  for (int i = 1; i < 8; ++i) bm = fmaxf(bm, red[i]);
  __syncthreads();
  float sum = 0.f;
#pragma unroll
  for (int i = 0; i < 8; ++i) { vals[i] = __expf(vals[i] - bm); sum += vals[i]; }
#pragma unroll
  for (int o = 16; o > 0; o >>= 1) sum += __shfl_xor(sum, o, 32);
  if (lane == 0) red[wid] = sum;
  __syncthreads();
  float tot = 0.f;
#pragma unroll
  for (int i = 0; i < 8; ++i) tot += red[i];
  const float inv = 1.f / tot;
#pragma unroll
  for (int i = 0; i < 8; ++i) {
    const float pr = vals[i] * inv;
    p[t + i * 256] = pr;
    ph[row * (size_t)kS + t + i * 256] = (_Float16)pr;
  }
}

// ---------------- Context: ctx(f32)[b,q,d] = sum_k P[b,q,k] * V[b,k,d]
__global__ void __launch_bounds__(256)
attn_context_kernel(const _Float16* __restrict__ ph, const _Float16* __restrict__ vh,
                    float* __restrict__ ctx) {
  __shared__ __align__(16) _Float16 As[TM2 * PITCH];
  __shared__ __align__(16) _Float16 Bs[TN2 * PITCH];
  const int t = threadIdx.x;
  const int b  = blockIdx.z;
  const int n0 = blockIdx.x * TN2;  // d tile
  const int m0 = blockIdx.y * TM2;  // q tile
  const int lane = t & 31, w = t >> 5;
  const int l16 = lane & 15, hi = lane >> 4;
  const int wr = w >> 2, wc = w & 3;
  const int arow = t >> 2, ak0 = (t & 3) * 8;
  const int brow = t >> 1, bk0 = (t & 1) * 16;  // V^T staging: 1 half-row (16 halves)/thread
  v8f acc[2][2] = {};
  for (int k0 = 0; k0 < kS; k0 += TK) {
    // Async stage P tile rows (64 x 32 halves, contiguous in k)
    cp_async16(&As[arow * PITCH + ak0],
               ph + ((size_t)b * kS + m0 + arow) * kS + k0 + ak0);
    // V^T staged with VALU (transpose): Bs[nn][kk] = V[b, k0+kk, n0+nn]
#pragma unroll
    for (int i = 0; i < 16; ++i)
      Bs[brow * PITCH + bk0 + i] =
          vh[((size_t)b * kS + k0 + bk0 + i) * kD + n0 + brow];
    cp_async_wait();
    __syncthreads();
    mma_step2(As, Bs, wr, wc, l16, hi, acc);
    __syncthreads();
  }
#pragma unroll
  for (int nt = 0; nt < 2; ++nt) {
    const int d = n0 + wc * 32 + nt * 16 + l16;
#pragma unroll
    for (int mt = 0; mt < 2; ++mt) {
#pragma unroll
      for (int r = 0; r < 8; ++r) {
        const int q = m0 + wr * 32 + mt * 16 + hi * 8 + r;
        ctx[((size_t)b * kS + q) * kD + d] = acc[mt][nt][r];
      }
    }
  }
}

extern "C" void kernel_launch(void* const* d_in, const int* in_sizes, int n_in,
                              void* d_out, int out_size, void* d_ws, size_t ws_size,
                              hipStream_t stream) {
  (void)in_sizes; (void)n_in; (void)out_size; (void)ws_size;
  const float* query = (const float*)d_in[0];
  const float* key   = (const float*)d_in[1];
  const float* value = (const float*)d_in[2];
  const int*   mask  = (const int*)d_in[3];
  const float* Wq = (const float*)d_in[4];
  const float* bq = (const float*)d_in[5];
  const float* Wk = (const float*)d_in[6];
  const float* bk = (const float*)d_in[7];
  const float* Wv = (const float*)d_in[8];
  const float* bv = (const float*)d_in[9];

  // Workspace: q/k/v f16 (3 x 16 MiB) + probabilities f16 (64 MiB) = 112 MiB
  _Float16* qh = (_Float16*)d_ws;
  _Float16* kh = qh + (size_t)kB * kS * kD;
  _Float16* vh = kh + (size_t)kB * kS * kD;
  _Float16* ph = vh + (size_t)kB * kS * kD;

  float* ctx = (float*)d_out;                  // [B,S,D]
  float* wgt = ctx + (size_t)kB * kS * kD;     // [B,S,S]

  const dim3 blk(256);
  const dim3 pg(kD / TN, (kB * kS) / TM);
  attn_proj_kernel<<<pg, blk, 0, stream>>>(query, Wq, bq, qh);
  attn_proj_kernel<<<pg, blk, 0, stream>>>(key,   Wk, bk, kh);
  attn_proj_kernel<<<pg, blk, 0, stream>>>(value, Wv, bv, vh);

  const int npairs = kB * kS * (kRot / 2);
  attn_rotary_kernel<<<(npairs + 255) / 256, 256, 0, stream>>>(qh);
  attn_rotary_kernel<<<(npairs + 255) / 256, 256, 0, stream>>>(kh);

  attn_scores_kernel<<<dim3(kS / TN2, kS / TM2, kB), blk, 0, stream>>>(qh, kh, mask, wgt);
  attn_softmax_kernel<<<kB * kS, 256, 0, stream>>>(wgt, ph);
  attn_context_kernel<<<dim3(kD / TN2, kS / TM2, kB), blk, 0, stream>>>(ph, vh, ctx);
}
